// MoeGateDecoder_85469849191062
// MI455X (gfx1250) — compile-verified, matrix operand order checked by reading, not run on hardware
//
#include <hip/hip_runtime.h>
#include <hip/hip_bf16.h>
#include <stdint.h>

typedef __attribute__((ext_vector_type(16))) _Float16 v16h;
typedef __attribute__((ext_vector_type(8)))  float    v8f;

#define LATENT 64
#define OUTCH 207
#define GATEIN 128

// ---------------------------------------------------------------------------
// CDNA5 16-bit A layout (16x32, MxK): lane owns row M = lane&15 (both halves);
//   VGPR v<4 : K = 8*half + 2*v + {0,1}        -> 16B chunk @ byte 16*half
//   VGPR v>=4: K = 16 + 8*half + 2*(v-4)+{0,1} -> 16B chunk @ byte 32+16*half
// B (32x16 KxN) mirrors this with lane owning column n = lane&15, so a
// transposed weight WT[o][k] loads with the identical address pattern.
// ---------------------------------------------------------------------------
struct AF { union { v16h h; uint32_t u[8]; }; };

__device__ __forceinline__ AF load_frag(const _Float16* __restrict__ p,
                                        int row, int ld, int kbase, int half) {
    AF f;
    const _Float16* r = p + (size_t)row * ld + kbase + 8 * half;
#pragma unroll
    for (int v = 0; v < 4; ++v) f.u[v]     = *(const uint32_t*)(r + 2 * v);
#pragma unroll
    for (int v = 0; v < 4; ++v) f.u[4 + v] = *(const uint32_t*)(r + 16 + 2 * v);
    return f;
}

__device__ __forceinline__ AF load_fragB_lds(const _Float16* base, int half) {
    AF f;
    const uint32_t* p0 = (const uint32_t*)(base + 8 * half);
    const uint32_t* p1 = (const uint32_t*)(base + 16 + 8 * half);
#pragma unroll
    for (int v = 0; v < 4; ++v) { f.u[v] = p0[v]; f.u[4 + v] = p1[v]; }
    return f;
}

// LDS B tile: 128 cols x 32 k, row padded to 40 f16 (80B) to stagger banks.
#define BROW 40
#define BUFB (128 * BROW * 2)   // 10240 bytes per buffer

// Async-DMA one 32k x 128col B chunk (8KB) into LDS: 512 x 16B segments,
// 2 per thread. CDNA5 GLOBAL_LOAD_ASYNC_TO_LDS_B128, tracked by ASYNCcnt.
__device__ __forceinline__ void stage_async(const _Float16* __restrict__ WT,
                                            int ldw, int col0, int kglob,
                                            uint32_t ldsbase) {
#pragma unroll
    for (int rep = 0; rep < 2; ++rep) {
        int s  = (int)threadIdx.x + rep * 256;
        int o  = s >> 2;
        int kk = (s & 3) * 8;
        const _Float16* g = WT + (size_t)(col0 + o) * ldw + kglob + kk;
        uint32_t lofs = ldsbase + (uint32_t)(o * (BROW * 2) + kk * 2);
        asm volatile("global_load_async_to_lds_b128 %0, %1, off"
                     :: "v"(lofs), "v"((unsigned long long)(uintptr_t)g)
                     : "memory");
    }
}
#define S_WAIT_ASYNC(IMM) asm volatile("s_wait_asynccnt " IMM ::: "memory")

// ---------------------------------------------------------------------------
// Fused (MoE-)GEMM:  Y[n,o] = act( sum_e sum_i coeff[n,e]*X[n,i]*W[e,i,o]
//                                  + sum_e coeff[n,e]*bias[e,o] )
// WT[o][e*KPER+i] f16. Wave tile 32x128 (2x8 subtiles), 8 waves/block ->
// block tile 256x128. B double-buffered in LDS via async DMA (shared by all
// 8 waves); A fragments double-buffered in *named* registers (rotation +
// unroll-2, no runtime indexing -> no scratch); B LDS fragments pipelined
// TWO ahead in explicit slots so ds_load latency hides under 4 WMMAs.
// ---------------------------------------------------------------------------
template<int KPER, int NEXP, bool DO_ELU, bool F16OUT>
__global__ void __launch_bounds__(256)
moe_gemm_wmma(const _Float16* __restrict__ X, int ldx,
              const _Float16* __restrict__ WT,
              const float* __restrict__ coeff,   // [N,8] (unused if NEXP==1)
              const float* __restrict__ bias,    // [NEXP, outReal]
              int outReal,
              void* __restrict__ Yout, int ldy) {
    __shared__ _Float16 Bs[2][128][BROW];

    const int lane = threadIdx.x & 31;
    const int wave = threadIdx.x >> 5;
    const int half = lane >> 4;
    const int lm   = lane & 15;
    const int row0 = (blockIdx.x * 8 + wave) * 32;
    const int col0 = blockIdx.y * 128;
    const int ldw  = NEXP * KPER;
    const uint32_t ldsb0 = (uint32_t)(size_t)&Bs[0][0][0];
    const _Float16* bsrow = &Bs[0][0][0] + (size_t)lm * BROW;  // + buf*BUFB/2 + ct*16*BROW

    v8f acc[2][8] = {};

    const int C = NEXP * KPER / 32;   // 4 / 80 / 144 / 128 : always even

    // prologue: stage chunk 0 (B->LDS) and prefetch chunk 0 A fragments
    stage_async(WT, ldw, col0, 0, ldsb0);
    AF a0c = load_frag(X, row0 + lm,      ldx, 0, half);
    AF a1c = load_frag(X, row0 + 16 + lm, ldx, 0, half);
    AF a0n, a1n;

#pragma unroll 2
    for (int c = 0; c < C; ++c) {
        const int buf = c & 1;                     // constant under unroll-2
        const int e   = (c * 32) / KPER;           // expert of this chunk

        if (c + 1 < C) {
            // issue next B stage + next A prefetch before draining current
            const int kg2 = (c + 1) * 32;
            stage_async(WT, ldw, col0, kg2, ldsb0 + (uint32_t)(((c + 1) & 1) * BUFB));
            const int kx2 = kg2 - (kg2 / KPER) * KPER;
            a0n = load_frag(X, row0 + lm,      ldx, kx2, half);
            a1n = load_frag(X, row0 + 16 + lm, ldx, kx2, half);
            S_WAIT_ASYNC("0x2");                   // oldest stage (c) done
        } else {
            S_WAIT_ASYNC("0x0");
        }
        __syncthreads();

        v16h a0 = a0c.h, a1 = a1c.h;
        if (NEXP > 1) {
            _Float16 c0 = (_Float16)coeff[(size_t)(row0 + lm) * 8 + e];
            _Float16 c1 = (_Float16)coeff[(size_t)(row0 + 16 + lm) * 8 + e];
            a0 = a0 * c0;
            a1 = a1 * c1;
        }

        const _Float16* brow = bsrow + buf * (BUFB / 2);
        // two-slot B pipeline: consume (b0,b1) while (nf0,nf1) are in flight
        AF b0 = load_fragB_lds(brow, half);
        AF b1 = load_fragB_lds(brow + 16 * BROW, half);
#pragma unroll
        for (int ct = 0; ct < 8; ct += 2) {
            AF nf0, nf1;
            if (ct + 2 < 8) {
                nf0 = load_fragB_lds(brow + (ct + 2) * 16 * BROW, half);
                nf1 = load_fragB_lds(brow + (ct + 3) * 16 * BROW, half);
            }
            acc[0][ct]     = __builtin_amdgcn_wmma_f32_16x16x32_f16(
                false, a0, false, b0.h, (short)0, acc[0][ct],     false, false);
            acc[1][ct]     = __builtin_amdgcn_wmma_f32_16x16x32_f16(
                false, a1, false, b0.h, (short)0, acc[1][ct],     false, false);
            acc[0][ct + 1] = __builtin_amdgcn_wmma_f32_16x16x32_f16(
                false, a0, false, b1.h, (short)0, acc[0][ct + 1], false, false);
            acc[1][ct + 1] = __builtin_amdgcn_wmma_f32_16x16x32_f16(
                false, a1, false, b1.h, (short)0, acc[1][ct + 1], false, false);
            if (ct + 2 < 8) { b0 = nf0; b1 = nf1; }
        }
        __syncthreads();

        a0c = a0n;                                 // rotation, folds under unroll-2
        a1c = a1n;
    }

    // Epilogue: blended bias + ELU + store.
    float bb[8][NEXP];
#pragma unroll
    for (int ct = 0; ct < 8; ++ct) {
        const int col = col0 + ct * 16 + lm;
#pragma unroll
        for (int e = 0; e < NEXP; ++e)
            bb[ct][e] = (col < outReal) ? bias[(size_t)e * outReal + col] : 0.0f;
    }

#pragma unroll
    for (int rt = 0; rt < 2; ++rt) {
#pragma unroll
        for (int v = 0; v < 8; ++v) {
            const int row = row0 + rt * 16 + v + 8 * half;
            float cf8[NEXP];
            if (NEXP > 1) {
                const float* cr = coeff + (size_t)row * 8;
#pragma unroll
                for (int e = 0; e < NEXP; ++e) cf8[e] = cr[e];
            }
#pragma unroll
            for (int ct = 0; ct < 8; ++ct) {
                float s = acc[rt][ct][v];
                if (NEXP > 1) {
#pragma unroll
                    for (int e = 0; e < NEXP; ++e) s += cf8[e] * bb[ct][e];
                } else {
                    s += bb[ct][0];
                }
                if (DO_ELU) s = (s > 0.0f) ? s : (__expf(s) - 1.0f);
                const int col = col0 + ct * 16 + lm;
                if (F16OUT) {
                    ((_Float16*)Yout)[(size_t)row * ldy + col] = (_Float16)s;
                } else if (col < outReal) {
                    ((float*)Yout)[(size_t)row * ldy + col] = s;
                }
            }
        }
    }
}

// ---------------------------------------------------------------------------
// Prep: strided f32 -> f16 copy (concat builder)
// ---------------------------------------------------------------------------
__global__ void k_copy16(const float* __restrict__ src, int lds_, int cnt,
                         _Float16* __restrict__ dst, int ldd, int coff, int n) {
    int gid = blockIdx.x * blockDim.x + threadIdx.x;
    if (gid >= n * cnt) return;
    int r = gid / cnt, c = gid - r * cnt;
    dst[(size_t)r * ldd + coff + c] = (_Float16)src[(size_t)r * lds_ + c];
}

// Weight transpose+convert: w[E,K,OUT] f32 -> WT[OUTP][E*K] f16 (zero-pad rows)
__global__ void k_wtrans(const float* __restrict__ w, _Float16* __restrict__ WT,
                         int E, int K, int OUT, int OUTP) {
    int gid = blockIdx.x * blockDim.x + threadIdx.x;
    int tot = OUTP * E * K;
    if (gid >= tot) return;
    int ek = gid % (E * K);
    int o  = gid / (E * K);
    int e = ek / K, i = ek - e * K;
    float v = (o < OUT) ? w[((size_t)e * K + i) * OUT + o] : 0.0f;
    WT[(size_t)o * (E * K) + ek] = (_Float16)v;
}

// Gate layer 3 + softmax: tiny (128x8) per-sample GEMV, f32.
__global__ void k_gate3_softmax(const _Float16* __restrict__ H2,
                                const float* __restrict__ g3w,  // [128,8]
                                const float* __restrict__ g3b,  // [8]
                                float* __restrict__ coeff_ws,
                                float* __restrict__ coeff_out, int n) {
    int nn = blockIdx.x * blockDim.x + threadIdx.x;
    if (nn >= n) return;
    const _Float16* h = H2 + (size_t)nn * GATEIN;
    float l[8];
#pragma unroll
    for (int j = 0; j < 8; ++j) l[j] = g3b[j];
    for (int i = 0; i < GATEIN; ++i) {
        float hv = (float)h[i];
#pragma unroll
        for (int j = 0; j < 8; ++j) l[j] += hv * g3w[i * 8 + j];
    }
    float m = l[0];
#pragma unroll
    for (int j = 1; j < 8; ++j) m = fmaxf(m, l[j]);
    float s = 0.0f;
#pragma unroll
    for (int j = 0; j < 8; ++j) { l[j] = __expf(l[j] - m); s += l[j]; }
    float inv = 1.0f / s;
#pragma unroll
    for (int j = 0; j < 8; ++j) {
        float p = l[j] * inv;
        coeff_ws[(size_t)nn * 8 + j]  = p;
        coeff_out[(size_t)nn * 8 + j] = p;
    }
}

// ---------------------------------------------------------------------------
extern "C" void kernel_launch(void* const* d_in, const int* in_sizes, int n_in,
                              void* d_out, int out_size, void* d_ws, size_t ws_size,
                              hipStream_t stream) {
    const float* latent = (const float*)d_in[0];
    const float* cond   = (const float*)d_in[1];
    const float* phase  = (const float*)d_in[2];   // [N,32,2] contiguous == [N,64]
    const float* g1w = (const float*)d_in[3]; const float* g1b = (const float*)d_in[4];
    const float* g2w = (const float*)d_in[5]; const float* g2b = (const float*)d_in[6];
    const float* g3w = (const float*)d_in[7]; const float* g3b = (const float*)d_in[8];
    const float* w1 = (const float*)d_in[9];  const float* b1 = (const float*)d_in[10];
    const float* w2 = (const float*)d_in[11]; const float* b2 = (const float*)d_in[12];
    const float* w3 = (const float*)d_in[13]; const float* b3 = (const float*)d_in[14];

    const int N = in_sizes[0] / LATENT;   // 16384

    // ---- workspace carve-out (f16 activations + transposed f16 weights) ----
    char* w = (char*)d_ws;
    auto take = [&](size_t bytes) { char* p = w; w += (bytes + 255) & ~(size_t)255; return p; };
    _Float16* X1  = (_Float16*)take((size_t)N * 320 * 2);   // [cond|latent]
    _Float16* X2  = (_Float16*)take((size_t)N * 576 * 2);   // [h1|latent]
    _Float16* X3  = (_Float16*)take((size_t)N * 512 * 2);   // h2
    _Float16* G0  = (_Float16*)take((size_t)N * 128 * 2);   // [phase|latent]
    _Float16* H1  = (_Float16*)take((size_t)N * 128 * 2);
    _Float16* H2  = (_Float16*)take((size_t)N * 128 * 2);
    _Float16* GT1 = (_Float16*)take((size_t)128 * 128 * 2);
    _Float16* GT2 = (_Float16*)take((size_t)128 * 128 * 2);
    _Float16* WT1 = (_Float16*)take((size_t)512 * (8 * 320) * 2);
    _Float16* WT2 = (_Float16*)take((size_t)512 * (8 * 576) * 2);
    _Float16* WT3 = (_Float16*)take((size_t)256 * (8 * 512) * 2);
    float*    CF  = (float*)   take((size_t)N * 8 * 4);

    float* pred      = (float*)d_out;                      // [N,207]
    float* coeff_out = (float*)d_out + (size_t)N * OUTCH;  // [N,8]

    const int TB = 256;
    auto blocks = [&](long long t) { return (int)((t + TB - 1) / TB); };

    // ---- prep: concats (f32->f16) ----
    k_copy16<<<blocks((long long)N * 256), TB, 0, stream>>>(cond,   256, 256, X1, 320, 0,   N);
    k_copy16<<<blocks((long long)N * 64),  TB, 0, stream>>>(latent,  64,  64, X1, 320, 256, N);
    k_copy16<<<blocks((long long)N * 64),  TB, 0, stream>>>(latent,  64,  64, X2, 576, 512, N);
    k_copy16<<<blocks((long long)N * 64),  TB, 0, stream>>>(phase,   64,  64, G0, 128, 0,   N);
    k_copy16<<<blocks((long long)N * 64),  TB, 0, stream>>>(latent,  64,  64, G0, 128, 64,  N);

    // ---- prep: weight transpose+convert ----
    k_wtrans<<<blocks(128LL * 128), TB, 0, stream>>>(g1w, GT1, 1, 128, 128, 128);
    k_wtrans<<<blocks(128LL * 128), TB, 0, stream>>>(g2w, GT2, 1, 128, 128, 128);
    k_wtrans<<<blocks(512LL * 8 * 320), TB, 0, stream>>>(w1, WT1, 8, 320, 512, 512);
    k_wtrans<<<blocks(512LL * 8 * 576), TB, 0, stream>>>(w2, WT2, 8, 576, 512, 512);
    k_wtrans<<<blocks(256LL * 8 * 512), TB, 0, stream>>>(w3, WT3, 8, 512, 207, 256);

    // ---- gate MLP (WMMA GEMMs, E=1, 128 cols per block) ----
    moe_gemm_wmma<128, 1, true, true><<<dim3(N / 256, 1), 256, 0, stream>>>(
        G0, 128, GT1, nullptr, g1b, 128, H1, 128);
    moe_gemm_wmma<128, 1, true, true><<<dim3(N / 256, 1), 256, 0, stream>>>(
        H1, 128, GT2, nullptr, g2b, 128, H2, 128);
    k_gate3_softmax<<<blocks(N), TB, 0, stream>>>(H2, g3w, g3b, CF, coeff_out, N);

    // ---- MoE trunk: single GEMM per layer, K' = 8*K, coeff folded into A ----
    moe_gemm_wmma<320, 8, true, true><<<dim3(N / 256, 4), 256, 0, stream>>>(
        X1, 320, WT1, CF, b1, 512, X2, 576);           // h1 into X2[:, :512]
    moe_gemm_wmma<576, 8, true, true><<<dim3(N / 256, 4), 256, 0, stream>>>(
        X2, 576, WT2, CF, b2, 512, X3, 512);
    moe_gemm_wmma<512, 8, false, false><<<dim3(N / 256, 2), 256, 0, stream>>>(
        X3, 512, WT3, CF, b3, OUTCH, pred, OUTCH);     // f32 out, cols >=207 masked
}
